// EV_GCN_85864986181827
// MI455X (gfx1250) — compile-verified, matrix operand order checked by reading, not run on hardware
//
#include <hip/hip_runtime.h>
#include <hip/hip_bf16.h>

#define NNODES 50000
#define NEDGES 1600000
#define DFEAT  128

typedef __attribute__((ext_vector_type(16))) _Float16 v16h;
typedef __attribute__((ext_vector_type(8)))  float    v8f;

__device__ __forceinline__ void wave_lds_sync() {
  // per-wave LDS ordering (all LDS tiles are wave-private): wait for DS ops
  asm volatile("s_wait_dscnt 0" ::: "memory");
}

__device__ __forceinline__ v8f wmma16x16x32(v16h a, v16h b, v8f c) {
  return __builtin_amdgcn_wmma_f32_16x16x32_f16(false, a, false, b, (short)0, c,
                                                false, false);
}

__device__ __forceinline__ void atomic_add_f32(float* p, float v) {
  unsafeAtomicAdd(p, v);  // native global_atomic_add_f32
}

// ---------------------------------------------------------------------------
// Weight packing into WMMA B-operand lane order (f16, 16x16x32 tiles).
// B[k][n] for a K x Ncol matrix; dst flat index:
//   (((kt*NT + nt)*32 + lane)*16 + h), NT = Ncol/16
//   k = kt*32 + (lane<16?0:8) + h + (h>=8?8:0),  n = nt*16 + (lane&15)
// ---------------------------------------------------------------------------
__device__ void pack_weight(_Float16* dst, const float* src, int K, int Ncol,
                            int tid, int nthreads) {
  const int NT = Ncol >> 4;
  const int total = K * Ncol;
  for (int idx = tid; idx < total; idx += nthreads) {
    int h = idx & 15;
    int lane = (idx >> 4) & 31;
    int t = idx >> 9;
    int nt = t % NT;
    int kt = t / NT;
    int koff = (lane < 16) ? 0 : 8;
    int k = kt * 32 + koff + h + ((h >= 8) ? 8 : 0);
    int n = nt * 16 + (lane & 15);
    dst[idx] = (_Float16)src[k * Ncol + n];
  }
}

__global__ void pack_kernel(
    const float* pae_g, const float* pae_b, const float* pae_rm, const float* pae_rv,
    float* pae_s, float* pae_t,
    const float* cls_g, const float* cls_b, const float* cls_rm, const float* cls_rv,
    float* cls_s, float* cls_t,
    const float* w2, _Float16* w2pk,
    const float* clsw1, _Float16* clspk,
    const float* ch0, _Float16* ch0pk,
    const float* ch1, _Float16* ch1pk,
    const float* ch2, _Float16* ch2pk,
    const float* ch3, _Float16* ch3pk) {
  const int tid = threadIdx.x;
  for (int i = tid; i < 128; i += 256) {
    float s = pae_g[i] * rsqrtf(pae_rv[i] + 1e-5f);
    pae_s[i] = s;
    pae_t[i] = pae_b[i] - pae_rm[i] * s;
  }
  for (int i = tid; i < 256; i += 256) {
    float s = cls_g[i] * rsqrtf(cls_rv[i] + 1e-5f);
    cls_s[i] = s;
    cls_t[i] = cls_b[i] - cls_rm[i] * s;
  }
  pack_weight(w2pk, w2, 128, 128, tid, 256);
  pack_weight(clspk, clsw1, 256, 256, tid, 256);
  for (int s = 0; s < 3; ++s)
    pack_weight(ch0pk + s * 128 * 64, ch0 + s * 128 * 64, 128, 64, tid, 256);
  for (int s = 0; s < 3; ++s)
    pack_weight(ch1pk + s * 64 * 64, ch1 + s * 64 * 64, 64, 64, tid, 256);
  for (int s = 0; s < 3; ++s)
    pack_weight(ch2pk + s * 64 * 64, ch2 + s * 64 * 64, 64, 64, tid, 256);
  for (int s = 0; s < 3; ++s)
    pack_weight(ch3pk + s * 64 * 64, ch3 + s * 64 * 64, 64, 64, tid, 256);
}

// ---------------------------------------------------------------------------
// PAE edge MLP: per wave, 16 edges. For each encoder half:
//   h = s*relu(x[:,6]@w1 + b1) + t  (BN folded), f16 -> LDS
//   out = h @ w2 + b2 via 8 ntiles x 4 ktiles of v_wmma_f32_16x16x32_f16
// then cosine similarity between the two 128-d outputs.
// ---------------------------------------------------------------------------
__global__ void __launch_bounds__(128)
pae_edge_kernel(const float* __restrict__ edgenet,
                const float* __restrict__ w1, const float* __restrict__ b1,
                const float* __restrict__ bns, const float* __restrict__ bnt,
                const _Float16* __restrict__ w2pack,
                const float* __restrict__ b2,
                float* __restrict__ ew_out) {
  __shared__ float   x_lds[4][16][12];
  __shared__ _Float16 h_lds[4][16][128];
  __shared__ float   o_lds[4][2][16][128];
  const int wid = threadIdx.x >> 5;
  const int lane = threadIdx.x & 31;
  const int tile = blockIdx.x * 4 + wid;
  const int e0 = tile * 16;

  {  // 16x12 inputs = 192 contiguous floats
    const float* src = edgenet + (long long)e0 * 12;
    float* dst = &x_lds[wid][0][0];
    for (int i = lane; i < 192; i += 32) dst[i] = src[i];
  }
  wave_lds_sync();

  const int rbase = (lane < 16) ? 0 : 8;
  const int koff = (lane < 16) ? 0 : 8;
  const int l16 = lane & 15;

  for (int enc = 0; enc < 2; ++enc) {
    {  // layer 1 (K=6) + folded BN, per lane: one half-row of h
      const int r = lane >> 1;
      const int c0 = (lane & 1) * 64;
      float xv[6];
#pragma unroll
      for (int d = 0; d < 6; ++d) xv[d] = x_lds[wid][r][enc * 6 + d];
      for (int j = 0; j < 64; ++j) {
        const int c = c0 + j;
        float sum = b1[c];
#pragma unroll
        for (int d = 0; d < 6; ++d) sum = fmaf(xv[d], w1[d * 128 + c], sum);
        h_lds[wid][r][c] = (_Float16)(bns[c] * fmaxf(sum, 0.f) + bnt[c]);
      }
    }
    wave_lds_sync();

#pragma unroll
    for (int nt = 0; nt < 8; ++nt) {
      v8f acc = {};
#pragma unroll
      for (int kt = 0; kt < 4; ++kt) {
        const _Float16* ap = &h_lds[wid][l16][kt * 32 + koff];
        v16h a;
#pragma unroll
        for (int j = 0; j < 8; ++j) { a[j] = ap[j]; a[8 + j] = ap[16 + j]; }
        const v16h b = *(const v16h*)(w2pack + (((kt * 8 + nt) * 32 + lane) << 4));
        acc = wmma16x16x32(a, b, acc);
      }
      const int col = nt * 16 + l16;
      const float bb = b2[col];
#pragma unroll
      for (int r = 0; r < 8; ++r) o_lds[wid][enc][rbase + r][col] = acc[r] + bb;
    }
    wave_lds_sync();
  }

  if (lane < 16) {  // cosine similarity per edge
    float dot = 0.f, n1 = 0.f, n2 = 0.f;
    const float* p1 = &o_lds[wid][0][lane][0];
    const float* p2 = &o_lds[wid][1][lane][0];
    for (int j = 0; j < 128; j += 4) {
      float4 a = *(const float4*)(p1 + j);
      float4 b = *(const float4*)(p2 + j);
      dot = fmaf(a.x, b.x, fmaf(a.y, b.y, fmaf(a.z, b.z, fmaf(a.w, b.w, dot))));
      n1 = fmaf(a.x, a.x, fmaf(a.y, a.y, fmaf(a.z, a.z, fmaf(a.w, a.w, n1))));
      n2 = fmaf(b.x, b.x, fmaf(b.y, b.y, fmaf(b.z, b.z, fmaf(b.w, b.w, n2))));
    }
    const float den = fmaxf(sqrtf(n1) * sqrtf(n2), 1e-8f);
    ew_out[e0 + lane] = (dot / den + 1.f) * 0.5f;
  }
}

// ---------------------------------------------------------------------------
// Graph plumbing kernels
// ---------------------------------------------------------------------------
__global__ void zero_kernel(float* __restrict__ p, long long n) {
  long long i = (long long)blockIdx.x * blockDim.x + threadIdx.x;
  if (i < n) p[i] = 0.f;
}

__global__ void deg_scatter_kernel(const float* __restrict__ ew,
                                   const int* __restrict__ row,
                                   float* __restrict__ deg, int e) {
  int i = blockIdx.x * blockDim.x + threadIdx.x;
  if (i < e) atomic_add_f32(&deg[row[i]], ew[i]);
}

__global__ void dis_kernel(float* __restrict__ deg, int n) {
  int i = blockIdx.x * blockDim.x + threadIdx.x;
  if (i < n) {
    float d = deg[i];
    deg[i] = (d > 0.f) ? rsqrtf(d) : 0.f;
  }
}

__global__ void norm_kernel(const float* __restrict__ ew,
                            const int* __restrict__ row,
                            const int* __restrict__ col,
                            const float* __restrict__ dis,
                            float* __restrict__ nrm, int e) {
  int i = blockIdx.x * blockDim.x + threadIdx.x;
  if (i < e) nrm[i] = -dis[row[i]] * ew[i] * dis[col[i]];
}

// T[row[e], :] += norm[e] * X[col[e], :]; one thread per (edge, 4 columns)
__global__ void lhat_scatter_kernel(const float* __restrict__ X, int ldx, int C,
                                    float* __restrict__ T,
                                    const float* __restrict__ nrm,
                                    const int* __restrict__ row,
                                    const int* __restrict__ col, int e) {
  long long tid = (long long)blockIdx.x * blockDim.x + threadIdx.x;
  const int cq = C >> 2;
  if (tid >= (long long)e * cq) return;
  const int ei = (int)(tid / cq);
  const int q = ((int)(tid % cq)) << 2;
  const float w = nrm[ei];
  const float4 v = *(const float4*)(X + (long long)col[ei] * ldx + q);
  float* tp = T + (long long)row[ei] * C + q;
  atomic_add_f32(tp + 0, w * v.x);
  atomic_add_f32(tp + 1, w * v.y);
  atomic_add_f32(tp + 2, w * v.z);
  atomic_add_f32(tp + 3, w * v.w);
}

// T2 = 2*T2 - X  (Tx2 = 2*lhat(Tx1) - Tx0)
__global__ void combine_kernel(float* __restrict__ T2, const float* __restrict__ X,
                               int ldx, int C, long long total) {
  long long i = (long long)blockIdx.x * blockDim.x + threadIdx.x;
  if (i >= total) return;
  long long n = i / C;
  int c = (int)(i % C);
  T2[i] = 2.f * T2[i] - X[n * ldx + c];
}

// ---------------------------------------------------------------------------
// ChebConv GEMM: out[:, colbase:colbase+64] = relu(X0@W0 + X1@W1 + X2@W2)
// per wave: 16 rows; activations converted f32->f16 into wave-private LDS.
// ---------------------------------------------------------------------------
__global__ void __launch_bounds__(128)
cheb_gemm_kernel(const float* __restrict__ X0, int ld0,
                 const float* __restrict__ X1, int ld1,
                 const float* __restrict__ X2, int ld2, int K,
                 const _Float16* __restrict__ wpack,  // [3][K*64] packed
                 float* __restrict__ out, int ldo, int colbase, int ntiles) {
  __shared__ _Float16 a_lds[4][3][16][128];
  const int wid = threadIdx.x >> 5;
  const int lane = threadIdx.x & 31;
  const int tile = blockIdx.x * 4 + wid;
  if (tile >= ntiles) return;

  const int row = lane & 15;
  const int half = lane >> 4;
  const long long grow = (long long)tile * 16 + row;
  const float* Xs[3] = {X0, X1, X2};
  const int lds[3] = {ld0, ld1, ld2};
  const int Kh = K >> 1;
#pragma unroll
  for (int s = 0; s < 3; ++s) {
    const float* xp = Xs[s] + grow * lds[s] + half * Kh;
    _Float16* ap = &a_lds[wid][s][row][half * Kh];
    for (int j = 0; j < Kh; j += 4) {
      float4 v = *(const float4*)(xp + j);
      ap[j] = (_Float16)v.x; ap[j + 1] = (_Float16)v.y;
      ap[j + 2] = (_Float16)v.z; ap[j + 3] = (_Float16)v.w;
    }
  }
  wave_lds_sync();

  const int KT = K >> 5;
  const int rbase = (lane < 16) ? 0 : 8;
  const int koff = (lane < 16) ? 0 : 8;
  const int l16 = lane & 15;
#pragma unroll
  for (int nt = 0; nt < 4; ++nt) {
    v8f acc = {};
#pragma unroll
    for (int s = 0; s < 3; ++s) {
      const _Float16* wp = wpack + s * (K * 64);
      for (int kt = 0; kt < KT; ++kt) {
        const _Float16* ap = &a_lds[wid][s][l16][kt * 32 + koff];
        v16h a;
#pragma unroll
        for (int j = 0; j < 8; ++j) { a[j] = ap[j]; a[8 + j] = ap[16 + j]; }
        const v16h b = *(const v16h*)(wp + (((kt * 4 + nt) * 32 + lane) << 4));
        acc = wmma16x16x32(a, b, acc);
      }
    }
    const int col = colbase + nt * 16 + l16;
#pragma unroll
    for (int r = 0; r < 8; ++r)
      out[((long long)tile * 16 + rbase + r) * ldo + col] = fmaxf(acc[r], 0.f);
  }
}

// ---------------------------------------------------------------------------
// Classifier head: z = bn(relu(jk@W1+b1)) (folded), logit = z@W2 + b2, fused.
// ---------------------------------------------------------------------------
__global__ void __launch_bounds__(128)
classifier_kernel(const float* __restrict__ jk,
                  const _Float16* __restrict__ wpack,  // 256x256 packed
                  const float* __restrict__ b1,
                  const float* __restrict__ bns, const float* __restrict__ bnt,
                  const float* __restrict__ w2, const float* __restrict__ b2,
                  float* __restrict__ logit, int ntiles) {
  __shared__ _Float16 a_lds[4][16][256];
  __shared__ float   z_lds[4][16][256];
  const int wid = threadIdx.x >> 5;
  const int lane = threadIdx.x & 31;
  const int tile = blockIdx.x * 4 + wid;
  if (tile >= ntiles) return;

  const int row = lane & 15;
  const int half = lane >> 4;
  {
    const float* xp = jk + ((long long)tile * 16 + row) * 256 + half * 128;
    _Float16* ap = &a_lds[wid][row][half * 128];
    for (int j = 0; j < 128; j += 4) {
      float4 v = *(const float4*)(xp + j);
      ap[j] = (_Float16)v.x; ap[j + 1] = (_Float16)v.y;
      ap[j + 2] = (_Float16)v.z; ap[j + 3] = (_Float16)v.w;
    }
  }
  wave_lds_sync();

  const int rbase = (lane < 16) ? 0 : 8;
  const int koff = (lane < 16) ? 0 : 8;
  const int l16 = lane & 15;
  for (int nt = 0; nt < 16; ++nt) {
    v8f acc = {};
#pragma unroll
    for (int kt = 0; kt < 8; ++kt) {
      const _Float16* ap = &a_lds[wid][l16][kt * 32 + koff];
      v16h a;
#pragma unroll
      for (int j = 0; j < 8; ++j) { a[j] = ap[j]; a[8 + j] = ap[16 + j]; }
      const v16h b = *(const v16h*)(wpack + (((kt * 16 + nt) * 32 + lane) << 4));
      acc = wmma16x16x32(a, b, acc);
    }
    const int col = nt * 16 + l16;
    const float s = bns[col], t = bnt[col], bb = b1[col];
#pragma unroll
    for (int r = 0; r < 8; ++r)
      z_lds[wid][rbase + r][col] = s * fmaxf(acc[r] + bb, 0.f) + t;
  }
  wave_lds_sync();

  if (lane < 16) {
    float a0 = b2[0], a1 = b2[1];
    const float* zp = &z_lds[wid][lane][0];
    for (int j = 0; j < 256; ++j) {
      float z = zp[j];
      a0 = fmaf(z, w2[2 * j + 0], a0);
      a1 = fmaf(z, w2[2 * j + 1], a1);
    }
    const long long g = (long long)tile * 16 + lane;
    logit[g * 2 + 0] = a0;
    logit[g * 2 + 1] = a1;
  }
}

// ---------------------------------------------------------------------------
// Host side
// ---------------------------------------------------------------------------
static constexpr size_t A256(size_t x) { return (x + 255) & ~(size_t)255; }
static constexpr size_t OFF_NORM = 0;
static constexpr size_t OFF_DEG  = A256(OFF_NORM + (size_t)NEDGES * 4);
static constexpr size_t OFF_PAES = A256(OFF_DEG + (size_t)NNODES * 4);
static constexpr size_t OFF_PAET = OFF_PAES + 512;
static constexpr size_t OFF_CLSS = OFF_PAET + 512;
static constexpr size_t OFF_CLST = OFF_CLSS + 1024;
static constexpr size_t OFF_W2PK = A256(OFF_CLST + 1024);
static constexpr size_t OFF_CLSPK = OFF_W2PK + 128 * 128 * 2;
static constexpr size_t OFF_CH0PK = OFF_CLSPK + 256 * 256 * 2;
static constexpr size_t OFF_CH1PK = OFF_CH0PK + 3 * 128 * 64 * 2;
static constexpr size_t OFF_CH2PK = OFF_CH1PK + 3 * 64 * 64 * 2;
static constexpr size_t OFF_CH3PK = OFF_CH2PK + 3 * 64 * 64 * 2;
static constexpr size_t OFF_T1 = A256(OFF_CH3PK + 3 * 64 * 64 * 2);
static constexpr size_t OFF_T2 = OFF_T1 + (size_t)NNODES * 128 * 4;

extern "C" void kernel_launch(void* const* d_in, const int* in_sizes, int n_in,
                              void* d_out, int out_size, void* d_ws, size_t ws_size,
                              hipStream_t stream) {
  (void)in_sizes; (void)n_in; (void)out_size; (void)ws_size;
  const float* features = (const float*)d_in[0];
  const int*   edge_index = (const int*)d_in[1];
  const float* edgenet = (const float*)d_in[2];
  const float* pae_w1 = (const float*)d_in[3];
  const float* pae_b1 = (const float*)d_in[4];
  const float* pae_g  = (const float*)d_in[5];
  const float* pae_b  = (const float*)d_in[6];
  const float* pae_rm = (const float*)d_in[7];
  const float* pae_rv = (const float*)d_in[8];
  const float* pae_w2 = (const float*)d_in[9];
  const float* pae_b2 = (const float*)d_in[10];
  const float* cheb_w0 = (const float*)d_in[11];
  const float* cheb_w1 = (const float*)d_in[12];
  const float* cheb_w2 = (const float*)d_in[13];
  const float* cheb_w3 = (const float*)d_in[14];
  const float* cls_w1 = (const float*)d_in[15];
  const float* cls_b1 = (const float*)d_in[16];
  const float* cls_g  = (const float*)d_in[17];
  const float* cls_b  = (const float*)d_in[18];
  const float* cls_rm = (const float*)d_in[19];
  const float* cls_rv = (const float*)d_in[20];
  const float* cls_w2 = (const float*)d_in[21];
  const float* cls_b2 = (const float*)d_in[22];

  const int* row = edge_index;
  const int* col = edge_index + NEDGES;

  char* ws = (char*)d_ws;
  float* nrm   = (float*)(ws + OFF_NORM);
  float* deg   = (float*)(ws + OFF_DEG);
  float* paeS  = (float*)(ws + OFF_PAES);
  float* paeT  = (float*)(ws + OFF_PAET);
  float* clsS  = (float*)(ws + OFF_CLSS);
  float* clsT  = (float*)(ws + OFF_CLST);
  _Float16* w2pk  = (_Float16*)(ws + OFF_W2PK);
  _Float16* clspk = (_Float16*)(ws + OFF_CLSPK);
  _Float16* ch0pk = (_Float16*)(ws + OFF_CH0PK);
  _Float16* ch1pk = (_Float16*)(ws + OFF_CH1PK);
  _Float16* ch2pk = (_Float16*)(ws + OFF_CH2PK);
  _Float16* ch3pk = (_Float16*)(ws + OFF_CH3PK);
  float* T1 = (float*)(ws + OFF_T1);
  float* T2 = (float*)(ws + OFF_T2);

  float* out_jk = (float*)d_out;                         // [N,256]
  float* out_logit = out_jk + (size_t)NNODES * 256;      // [N,2]
  float* out_ew = out_logit + (size_t)NNODES * 2;        // [E]

  // 1. pack weights + fold BN
  pack_kernel<<<1, 256, 0, stream>>>(pae_g, pae_b, pae_rm, pae_rv, paeS, paeT,
                                     cls_g, cls_b, cls_rm, cls_rv, clsS, clsT,
                                     pae_w2, w2pk, cls_w1, clspk,
                                     cheb_w0, ch0pk, cheb_w1, ch1pk,
                                     cheb_w2, ch2pk, cheb_w3, ch3pk);

  // 2. edge weights (WMMA)
  pae_edge_kernel<<<NEDGES / 16 / 4, 128, 0, stream>>>(
      edgenet, pae_w1, pae_b1, paeS, paeT, w2pk, pae_b2, out_ew);

  // 3. degree / normalization
  zero_kernel<<<(NNODES + 255) / 256, 256, 0, stream>>>(deg, NNODES);
  deg_scatter_kernel<<<NEDGES / 256, 256, 0, stream>>>(out_ew, row, deg, NEDGES);
  dis_kernel<<<(NNODES + 255) / 256, 256, 0, stream>>>(deg, NNODES);
  norm_kernel<<<NEDGES / 256, 256, 0, stream>>>(out_ew, row, col, deg, nrm, NEDGES);

  // 4. ChebConv stack with JK concat into out_jk
  const int NTILES = NNODES / 16;            // 3125
  const int GBLK = (NTILES + 3) / 4;         // 782
  struct Layer { const float* X; int ldx; int K; const _Float16* wpk; int cb; };
  Layer layers[4] = {
      {features,      128, 128, ch0pk, 0},
      {out_jk + 0,    256, 64,  ch1pk, 64},
      {out_jk + 64,   256, 64,  ch2pk, 128},
      {out_jk + 128,  256, 64,  ch3pk, 192},
  };
  for (int l = 0; l < 4; ++l) {
    const Layer& L = layers[l];
    const long long nK = (long long)NNODES * L.K;
    const long long sc = (long long)NEDGES * (L.K >> 2);
    // Tx1 = lhat(X)
    zero_kernel<<<(unsigned)((nK + 255) / 256), 256, 0, stream>>>(T1, nK);
    lhat_scatter_kernel<<<(unsigned)((sc + 255) / 256), 256, 0, stream>>>(
        L.X, L.ldx, L.K, T1, nrm, row, col, NEDGES);
    // Tx2 = 2*lhat(Tx1) - X
    zero_kernel<<<(unsigned)((nK + 255) / 256), 256, 0, stream>>>(T2, nK);
    lhat_scatter_kernel<<<(unsigned)((sc + 255) / 256), 256, 0, stream>>>(
        T1, L.K, L.K, T2, nrm, row, col, NEDGES);
    combine_kernel<<<(unsigned)((nK + 255) / 256), 256, 0, stream>>>(
        T2, L.X, L.ldx, L.K, nK);
    // h = relu(X@W0 + T1@W1 + T2@W2) -> jk columns [cb, cb+64)
    cheb_gemm_kernel<<<GBLK, 128, 0, stream>>>(L.X, L.ldx, T1, L.K, T2, L.K,
                                               L.K, L.wpk, out_jk, 256, L.cb,
                                               NTILES);
  }

  // 5. classifier head
  classifier_kernel<<<GBLK, 128, 0, stream>>>(out_jk, clspk, cls_b1, clsS, clsT,
                                              cls_w2, cls_b2, out_logit, NTILES);
}
